// LightGCN_34694745817427
// MI455X (gfx1250) — matrix operand order repaired and checked by reading, not compiled, not run on hardware
//
#include <hip/hip_runtime.h>
#include <hip/hip_bf16.h>

#define DEV __device__ __forceinline__

typedef __attribute__((ext_vector_type(16))) _Float16 v16h;
typedef __attribute__((ext_vector_type(8)))  float    v8f;

constexpr int N_USERS = 100000;
constexpr int N_ITEMS = 50000;
constexpr int N_NODES = N_USERS + N_ITEMS;
constexpr int D       = 64;
constexpr int NNZ     = 2000000;
constexpr int BATCH   = 256;
constexpr float LAYER_W = 0.25f;   // 1/(N_LAYERS+1)

// ---------------- WMMA helpers (gfx1250, wave32) ----------------

DEV v8f wmma_f16(v16h a, v16h b, v8f c) {
  // D = A(16x32 f16) * B(32x16 f16) + C(16x16 f32)
  return __builtin_amdgcn_wmma_f32_16x16x32_f16(false, a, false, b, (short)0, c,
                                                false, false);
}

// A-matrix loader, source row-major f32, per-lane row pointer.
// Layout (16-bit A 16x32, matching B): lanes 0-15 hold K=0..15,
// lanes 16-31 hold K=16..31; M = lane&15.
DEV v16h load_a_row_f32(const float* rowp, int kbase, int lane) {
  int c0 = ((lane & 16) ? 16 : 0) + kbase;
  v16h a;
#pragma unroll
  for (int j = 0; j < 8; ++j) a[j] = (_Float16)rowp[c0 + j];
#pragma unroll
  for (int j = 0; j < 8; ++j) a[8 + j] = (_Float16)rowp[c0 + 8 + j];
  return a;
}

DEV v16h load_a_row_h(const _Float16* rowp, int kbase, int lane) {
  int c0 = ((lane & 16) ? 16 : 0) + kbase;
  v16h a;
#pragma unroll
  for (int j = 0; j < 8; ++j) a[j] = rowp[c0 + j];
#pragma unroll
  for (int j = 0; j < 8; ++j) a[8 + j] = rowp[c0 + 8 + j];
  return a;
}

// B-matrix loader from Bt (= B transposed, N x K row-major f16).
// Layout (32x16 16-bit B): lane&15 = column n; lanes 0-15 hold K=0..15,
// lanes 16-31 hold K=16..31, sequential K within the 8 VGPRs.
DEV v16h load_bt_h(const _Float16* bt, int ldk, int nblk, int kbase, int lane) {
  const _Float16* p = bt + (size_t)(nblk * 16 + (lane & 15)) * ldk + kbase +
                      ((lane & 16) ? 16 : 0);
  v16h b;
#pragma unroll
  for (int i = 0; i < 16; ++i) b[i] = p[i];
  return b;
}

// C/D layout: VGPR j, lane l -> (M = j + 8*(l>=16), N = l&15)

// ---------------- prep: f16-transpose all dense weights ----------------

__global__ void k_prep_weights(const float* gW1, const float* iW1,
                               const float* iW2, const float* uW1,
                               const float* uW2, _Float16* gW1t,
                               _Float16* iW1t, _Float16* iW2t, _Float16* uW1t,
                               _Float16* uW2t) {
  int t = threadIdx.x;
  for (int i = t; i < 128 * 64; i += blockDim.x) {
    int k = i >> 6, n = i & 63;
    gW1t[n * 128 + k] = (_Float16)gW1[i];
  }
  for (int i = t; i < 64 * 64; i += blockDim.x) {
    int k = i >> 6, n = i & 63;
    iW1t[n * 64 + k] = (_Float16)iW1[i];
    iW2t[n * 64 + k] = (_Float16)iW2[i];
    uW1t[n * 64 + k] = (_Float16)uW1[i];
    uW2t[n * 64 + k] = (_Float16)uW2[i];
  }
}

// ---------------- graph propagation ----------------

__global__ void k_init(const float* ue, const float* ie, float* outA,
                       float* cur) {
  int i = blockIdx.x * blockDim.x + threadIdx.x;  // float4 index
  const int n4 = N_NODES * D / 4;
  if (i >= n4) return;
  const int u4 = N_USERS * D / 4;
  float4 v = (i < u4) ? ((const float4*)ue)[i] : ((const float4*)ie)[i - u4];
  ((float4*)cur)[i] = v;
  float4 o = {v.x * LAYER_W, v.y * LAYER_W, v.z * LAYER_W, v.w * LAYER_W};
  ((float4*)outA)[i] = o;
}

__global__ void k_zero(float* p, int n4) {
  int i = blockIdx.x * blockDim.x + threadIdx.x;
  if (i < n4) ((float4*)p)[i] = float4{0.f, 0.f, 0.f, 0.f};
}

__global__ void k_accum(float* outA, const float* nxt, int n4) {
  int i = blockIdx.x * blockDim.x + threadIdx.x;
  if (i >= n4) return;
  float4 o = ((float4*)outA)[i];
  float4 n = ((const float4*)nxt)[i];
  o.x += n.x * LAYER_W; o.y += n.y * LAYER_W;
  o.z += n.z * LAYER_W; o.w += n.w * LAYER_W;
  ((float4*)outA)[i] = o;
}

// One wave per edge; lane handles 2 of the 64 dims. Embedding table is
// L2-resident (38.4MB << 192MB L2) so gathers/atomics stay on-chip.
__global__ void k_spmm(const float* __restrict__ vals,
                       const int* __restrict__ rows,
                       const int* __restrict__ cols,
                       const float* __restrict__ cur, float* nxt) {
  int wid = (blockIdx.x * blockDim.x + threadIdx.x) >> 5;
  int lane = threadIdx.x & 31;
  if (wid >= NNZ) return;
  float v = vals[wid];
  int r = rows[wid];
  int c = cols[wid];
  float2 m = ((const float2*)(cur + (size_t)c * D))[lane];
  float* dst = nxt + (size_t)r * D + lane * 2;
  atomicAdd(dst, v * m.x);
  atomicAdd(dst + 1, v * m.y);
}

// ---------------- fused item pipeline: gate MLP -> fuse -> item MLP ------

__global__ void k_items(const float* __restrict__ outA,
                        const float* __restrict__ pop_emb,
                        const int* __restrict__ bins,
                        const _Float16* __restrict__ gW1t,
                        const float* __restrict__ gb1,
                        const float* __restrict__ gW2,
                        const float* __restrict__ gb2,
                        const _Float16* __restrict__ iW1t,
                        const float* __restrict__ ib1,
                        const _Float16* __restrict__ iW2t,
                        const float* __restrict__ ib2,
                        _Float16* __restrict__ Ih) {
  __shared__ float    hf[8][16 * D];      // f32 staging tile per wave
  __shared__ _Float16 fh[8][16 * D];      // f16 fused-items tile per wave
  __shared__ float    zb[8][16];          // gate z per item
  const int wslot = threadIdx.x >> 5;
  const int lane = threadIdx.x & 31;
  const int tile = blockIdx.x * 8 + wslot;           // wave-uniform
  if (tile >= N_ITEMS / 16) return;
  const int it0 = tile * 16;
  const int r = lane & 15;
  const int item = it0 + r;
  const int nloc = lane & 15;
  const int mb = (lane & 16) ? 8 : 0;
  const float* xrow = outA + (size_t)(N_USERS + item) * D;
  const float* prow = pop_emb + (size_t)bins[item] * D;
  float* Hf = hf[wslot];
  _Float16* Fh = fh[wslot];
  float* Z = zb[wslot];
  const v8f vzero = {};

  // ---- gate GEMM: h = relu([x|p] @ gW1 + gb1), K=128 ----
  v8f acc[4];
#pragma unroll
  for (int nt = 0; nt < 4; ++nt) acc[nt] = vzero;
#pragma unroll
  for (int kc = 0; kc < 4; ++kc) {
    v16h a = (kc < 2) ? load_a_row_f32(xrow, kc * 32, lane)
                      : load_a_row_f32(prow, (kc - 2) * 32, lane);
#pragma unroll
    for (int nt = 0; nt < 4; ++nt)
      acc[nt] = wmma_f16(a, load_bt_h(gW1t, 128, nt, kc * 32, lane), acc[nt]);
  }
#pragma unroll
  for (int nt = 0; nt < 4; ++nt) {
    float bias = gb1[nt * 16 + nloc];
#pragma unroll
    for (int j = 0; j < 8; ++j)
      Hf[(mb + j) * D + nt * 16 + nloc] = fmaxf(acc[nt][j] + bias, 0.f);
  }
  asm volatile("s_wait_dscnt 0" ::: "memory");

  // ---- z = sigmoid(h . gW2 + gb2); every lane computes its item's dot ----
  float dot = 0.f;
#pragma unroll
  for (int n = 0; n < D; ++n) dot += Hf[r * D + n] * gW2[n];
  float zg = 1.f / (1.f + __expf(-(dot + gb2[0])));
  Z[r] = zg;  // lanes r and r+16 write identical values
  asm volatile("s_wait_dscnt 0" ::: "memory");

  // ---- items_fused = (1-z)*x + z*p, as f16 tile in LDS ----
  for (int idx = lane; idx < 16 * D; idx += 32) {
    int rr = idx >> 6, d = idx & 63;
    int it = it0 + rr;
    float x = outA[(size_t)(N_USERS + it) * D + d];
    float p = pop_emb[(size_t)bins[it] * D + d];
    float zz = Z[rr];
    Fh[idx] = (_Float16)((1.f - zz) * x + zz * p);
  }
  asm volatile("s_wait_dscnt 0" ::: "memory");

  // ---- item MLP layer 1: h2 = relu(F @ iW1 + ib1) ----
#pragma unroll
  for (int nt = 0; nt < 4; ++nt) acc[nt] = vzero;
#pragma unroll
  for (int kc = 0; kc < 2; ++kc) {
    v16h a = load_a_row_h(Fh + r * D, kc * 32, lane);
#pragma unroll
    for (int nt = 0; nt < 4; ++nt)
      acc[nt] = wmma_f16(a, load_bt_h(iW1t, 64, nt, kc * 32, lane), acc[nt]);
  }
#pragma unroll
  for (int nt = 0; nt < 4; ++nt) {
    float bias = ib1[nt * 16 + nloc];
#pragma unroll
    for (int j = 0; j < 8; ++j)
      Hf[(mb + j) * D + nt * 16 + nloc] = fmaxf(acc[nt][j] + bias, 0.f);
  }
  asm volatile("s_wait_dscnt 0" ::: "memory");

  // ---- item MLP layer 2: I = h2 @ iW2 + ib2, store f16 ----
#pragma unroll
  for (int nt = 0; nt < 4; ++nt) acc[nt] = vzero;
#pragma unroll
  for (int kc = 0; kc < 2; ++kc) {
    v16h a = load_a_row_f32(Hf + r * D, kc * 32, lane);
#pragma unroll
    for (int nt = 0; nt < 4; ++nt)
      acc[nt] = wmma_f16(a, load_bt_h(iW2t, 64, nt, kc * 32, lane), acc[nt]);
  }
#pragma unroll
  for (int nt = 0; nt < 4; ++nt) {
    float bias = ib2[nt * 16 + nloc];
#pragma unroll
    for (int j = 0; j < 8; ++j)
      Ih[(size_t)(it0 + mb + j) * D + nt * 16 + nloc] =
          (_Float16)(acc[nt][j] + bias);
  }
}

// ---------------- user MLP (256 users) ----------------

__global__ void k_users(const float* __restrict__ outA,
                        const int* __restrict__ users,
                        const _Float16* __restrict__ uW1t,
                        const float* __restrict__ ub1,
                        const _Float16* __restrict__ uW2t,
                        const float* __restrict__ ub2,
                        _Float16* __restrict__ Uh) {
  __shared__ float hf[8][16 * D];
  const int wslot = threadIdx.x >> 5;
  const int lane = threadIdx.x & 31;
  const int r = lane & 15;
  const int nloc = lane & 15;
  const int mb = (lane & 16) ? 8 : 0;
  const v8f vzero = {};
  float* Hf = hf[wslot];
  for (int tile = wslot; tile < BATCH / 16; tile += 8) {
    int u = users[tile * 16 + r];
    const float* xrow = outA + (size_t)u * D;
    v8f acc[4];
#pragma unroll
    for (int nt = 0; nt < 4; ++nt) acc[nt] = vzero;
#pragma unroll
    for (int kc = 0; kc < 2; ++kc) {
      v16h a = load_a_row_f32(xrow, kc * 32, lane);
#pragma unroll
      for (int nt = 0; nt < 4; ++nt)
        acc[nt] = wmma_f16(a, load_bt_h(uW1t, 64, nt, kc * 32, lane), acc[nt]);
    }
#pragma unroll
    for (int nt = 0; nt < 4; ++nt) {
      float bias = ub1[nt * 16 + nloc];
#pragma unroll
      for (int j = 0; j < 8; ++j)
        Hf[(mb + j) * D + nt * 16 + nloc] = fmaxf(acc[nt][j] + bias, 0.f);
    }
    asm volatile("s_wait_dscnt 0" ::: "memory");
#pragma unroll
    for (int nt = 0; nt < 4; ++nt) acc[nt] = vzero;
#pragma unroll
    for (int kc = 0; kc < 2; ++kc) {
      v16h a = load_a_row_f32(Hf + r * D, kc * 32, lane);
#pragma unroll
      for (int nt = 0; nt < 4; ++nt)
        acc[nt] = wmma_f16(a, load_bt_h(uW2t, 64, nt, kc * 32, lane), acc[nt]);
    }
#pragma unroll
    for (int nt = 0; nt < 4; ++nt) {
      float bias = ub2[nt * 16 + nloc];
#pragma unroll
      for (int j = 0; j < 8; ++j)
        Uh[(size_t)(tile * 16 + mb + j) * D + nt * 16 + nloc] =
            (_Float16)(acc[nt][j] + bias);
    }
    asm volatile("s_wait_dscnt 0" ::: "memory");
  }
}

// ---------------- scores = U @ I^T + biases ----------------
// One wave per 16x16 output tile: A = 16 users (held in regs), B = I^T whose
// columns are contiguous I rows -> plain contiguous f16 loads give B layout.

__global__ void k_scores(const _Float16* __restrict__ Uh,
                         const _Float16* __restrict__ Ih,
                         const float* __restrict__ user_bias,
                         const float* __restrict__ item_bias,
                         const int* __restrict__ users,
                         float* __restrict__ out) {
  int wid = blockIdx.x * 8 + (threadIdx.x >> 5);
  int lane = threadIdx.x & 31;
  int nt = wid >> 4;  // item tile 0..3124
  int mt = wid & 15;  // user tile 0..15
  if (nt >= N_ITEMS / 16) return;
  int r = lane & 15;
  const _Float16* arow = Uh + (size_t)(mt * 16 + r) * D;
  v16h a0 = load_a_row_h(arow, 0, lane);
  v16h a1 = load_a_row_h(arow, 32, lane);

  int n0 = nt * 16;
  const _Float16* brow = Ih + (size_t)(n0 + r) * D + ((lane & 16) ? 16 : 0);
  __builtin_prefetch(Ih + (size_t)(n0 + 128 + r) * D, 0, 1);
  v16h b0, b1;
#pragma unroll
  for (int i = 0; i < 16; ++i) { b0[i] = brow[i]; b1[i] = brow[32 + i]; }

  v8f acc = {};
  acc = wmma_f16(a0, b0, acc);
  acc = wmma_f16(a1, b1, acc);

  int nloc = lane & 15;
  int mb = (lane & 16) ? 8 : 0;
  float ib = item_bias[n0 + nloc];
#pragma unroll
  for (int j = 0; j < 8; ++j) {
    int m = mt * 16 + mb + j;
    float ub = user_bias[users[m]];
    out[(size_t)m * N_ITEMS + n0 + nloc] = acc[j] + ub + ib;
  }
}

// ---------------- host-side orchestration ----------------

extern "C" void kernel_launch(void* const* d_in, const int* in_sizes, int n_in,
                              void* d_out, int out_size, void* d_ws,
                              size_t ws_size, hipStream_t stream) {
  const float* user_emb  = (const float*)d_in[0];
  const float* item_emb  = (const float*)d_in[1];
  const float* user_bias = (const float*)d_in[2];
  const float* item_bias = (const float*)d_in[3];
  const float* pop_emb   = (const float*)d_in[4];
  const float* uW1 = (const float*)d_in[5];
  const float* ub1 = (const float*)d_in[6];
  const float* uW2 = (const float*)d_in[7];
  const float* ub2 = (const float*)d_in[8];
  const float* iW1 = (const float*)d_in[9];
  const float* ib1 = (const float*)d_in[10];
  const float* iW2 = (const float*)d_in[11];
  const float* ib2 = (const float*)d_in[12];
  const float* gW1 = (const float*)d_in[13];
  const float* gb1 = (const float*)d_in[14];
  const float* gW2 = (const float*)d_in[15];
  const float* gb2 = (const float*)d_in[16];
  const float* adj_vals = (const float*)d_in[17];
  const int* adj_rows = (const int*)d_in[18];
  const int* adj_cols = (const int*)d_in[19];
  const int* bins  = (const int*)d_in[20];
  const int* users = (const int*)d_in[21];

  char* w = (char*)d_ws;
  auto alloc = [&](size_t bytes) {
    char* p = w;
    w += (bytes + 255) & ~size_t(255);
    return p;
  };
  const size_t ND = (size_t)N_NODES * D;
  float* outA = (float*)alloc(sizeof(float) * ND);
  float* cur  = (float*)alloc(sizeof(float) * ND);
  float* nxt  = (float*)alloc(sizeof(float) * ND);
  _Float16* Ih   = (_Float16*)alloc(sizeof(_Float16) * (size_t)N_ITEMS * D);
  _Float16* Uh   = (_Float16*)alloc(sizeof(_Float16) * BATCH * D);
  _Float16* gW1t = (_Float16*)alloc(sizeof(_Float16) * 64 * 128);
  _Float16* iW1t = (_Float16*)alloc(sizeof(_Float16) * 64 * 64);
  _Float16* iW2t = (_Float16*)alloc(sizeof(_Float16) * 64 * 64);
  _Float16* uW1t = (_Float16*)alloc(sizeof(_Float16) * 64 * 64);
  _Float16* uW2t = (_Float16*)alloc(sizeof(_Float16) * 64 * 64);

  k_prep_weights<<<1, 256, 0, stream>>>(gW1, iW1, iW2, uW1, uW2, gW1t, iW1t,
                                        iW2t, uW1t, uW2t);

  const int n4 = N_NODES * D / 4;
  const int blk4 = (n4 + 255) / 256;
  k_init<<<blk4, 256, 0, stream>>>(user_emb, item_emb, outA, cur);

  for (int l = 0; l < 3; ++l) {
    k_zero<<<blk4, 256, 0, stream>>>(nxt, n4);
    k_spmm<<<(NNZ + 7) / 8, 256, 0, stream>>>(adj_vals, adj_rows, adj_cols,
                                              cur, nxt);
    k_accum<<<blk4, 256, 0, stream>>>(outA, nxt, n4);
    float* t = cur; cur = nxt; nxt = t;
  }

  k_items<<<(N_ITEMS / 16 + 7) / 8, 256, 0, stream>>>(
      outA, pop_emb, bins, gW1t, gb1, gW2, gb2, iW1t, ib1, iW2t, ib2, Ih);
  k_users<<<1, 256, 0, stream>>>(outA, users, uW1t, ub1, uW2t, ub2, Uh);
  k_scores<<<(16 * (N_ITEMS / 16)) / 8, 256, 0, stream>>>(
      Uh, Ih, user_bias, item_bias, users, (float*)d_out);
}
